// SwinTransformerBlock4D_10067403342466
// MI455X (gfx1250) — matrix operand order repaired and checked by reading, not compiled, not run on hardware
//
#include <hip/hip_runtime.h>

typedef _Float16 f16;
typedef __attribute__((ext_vector_type(16))) _Float16 v16h;
typedef __attribute__((ext_vector_type(8)))  _Float16 h8;
typedef __attribute__((ext_vector_type(8)))  float    v8f;
typedef int i128v __attribute__((vector_size(16)));

#define WSZ    8
#define SHIFTV 4
#define HEADS  8
#define BATCH  4
#define CH     256
#define HH     256
#define WW_    256
#define HW     65536
#define NTOK   262144   /* BATCH*HW */
#define ATTN_SCALE 0.17677669529663687f
#define EPSV 1e-5f

// ---------------------------------------------------------------------------
// CDNA5 async global->LDS support (ASYNCcnt-tracked DMA), with sync fallback
// ---------------------------------------------------------------------------
#if defined(__has_builtin)
#  if __has_builtin(__builtin_amdgcn_global_load_async_to_lds_b128)
#    define HAVE_ASYNC_LDS 1
#  endif
#endif
#ifndef HAVE_ASYNC_LDS
#  define HAVE_ASYNC_LDS 0
#endif

__device__ __forceinline__ void async_cp16(const f16* g, f16* l) {
#if HAVE_ASYNC_LDS
    __builtin_amdgcn_global_load_async_to_lds_b128(
        (__attribute__((address_space(1))) i128v*)(unsigned long long)(uintptr_t)g,
        (__attribute__((address_space(3))) i128v*)(unsigned)(uintptr_t)l,
        0, 0);
#else
    *(uint4*)l = *(const uint4*)g;
#endif
}

template <int N>
__device__ __forceinline__ void wait_async() {
#if HAVE_ASYNC_LDS
#  if defined(__has_builtin) && __has_builtin(__builtin_amdgcn_s_wait_asynccnt)
    __builtin_amdgcn_s_wait_asynccnt((unsigned short)N);
#  else
    asm volatile("s_wait_asynccnt %0" :: "i"(N) : "memory");
#  endif
#endif
}

// ---------------------------------------------------------------------------
// K0: f32 -> f16 weight convert
// ---------------------------------------------------------------------------
__global__ void k_cvt(const float* __restrict__ a, f16* __restrict__ b, int n) {
    int i = blockIdx.x * 256 + threadIdx.x;
    if (i < n) b[i] = (f16)a[i];
}

// ---------------------------------------------------------------------------
// K1: LN1 + transpose (BCHW -> token-major shortcut f32) + cyclic shift +
//     window partition -> xw f16 (window-major rows, C contiguous)
// grid: B*H*(W/32) = 8192 blocks, 256 threads
// ---------------------------------------------------------------------------
__global__ __launch_bounds__(256) void k1_ln_shift(
    const float* __restrict__ x, const float* __restrict__ nw,
    const float* __restrict__ nb, float* __restrict__ shortcut,
    f16* __restrict__ xw)
{
    __shared__ float tile[256][33];
    __shared__ float red[2][8][32];
    __shared__ float mu_s[32], rs_s[32];

    int bid = blockIdx.x;
    int b   = bid >> 11;            // 2048 blocks per batch (H * W/32)
    int rem = bid & 2047;
    int h   = rem >> 3;
    int w0  = (rem & 7) << 5;
    int tid = threadIdx.x;
    int g = tid >> 5, l = tid & 31;

    const float* xb = x + (size_t)b * CH * HW + (size_t)h * WW_;
#pragma unroll
    for (int i = 0; i < 32; ++i) {
        int c = i * 8 + g;
        tile[c][l] = xb[(size_t)c * HW + w0 + l];
    }
    __syncthreads();

    float s = 0.f, s2 = 0.f;
#pragma unroll
    for (int i = 0; i < 32; ++i) {
        float v = tile[g * 32 + i][l];
        s += v; s2 += v * v;
    }
    red[0][g][l] = s; red[1][g][l] = s2;
    __syncthreads();
    if (tid < 32) {
        float sum = 0.f, sq = 0.f;
#pragma unroll
        for (int gg = 0; gg < 8; ++gg) { sum += red[0][gg][tid]; sq += red[1][gg][tid]; }
        float mu = sum * (1.0f / 256.0f);
        float var = sq * (1.0f / 256.0f) - mu * mu;
        mu_s[tid] = mu;
        rs_s[tid] = rsqrtf(var + EPSV);
    }
    __syncthreads();

    float wgt = nw[tid], bia = nb[tid];
#pragma unroll 4
    for (int tok = 0; tok < 32; ++tok) {
        float v = tile[tid][tok];
        size_t t = (size_t)b * HW + (size_t)h * WW_ + w0 + tok;
        shortcut[t * CH + tid] = v;                       // token-major residual
        float vn = (v - mu_s[tok]) * rs_s[tok] * wgt + bia;
        int hs = (h - SHIFTV) & 255;
        int ws = (w0 + tok - SHIFTV) & 255;
        int row = ((b * 32 + (hs >> 3)) * 32 + (ws >> 3)) * 64 + ((hs & 7) << 3) + (ws & 7);
        xw[(size_t)row * CH + tid] = (f16)vn;
    }
}

// ---------------------------------------------------------------------------
// Tiled WMMA GEMM: Y[m][n] = sum_k A[m][k] * W[n][k]  (+ per-mode epilogue)
// block = 256 threads (8 waves, 2x4), BM=BN=128, BK=32
// Double-buffered LDS filled by async global->LDS DMA (ASYNCcnt).
// MODE 0: +bias -> f16 out        (QKV)
// MODE 1: +bias +shortcut, window-reverse/unshift -> f32 x2 (proj)
// MODE 2: +bias, exact GELU -> f16 out (fc1)
// MODE 3: +bias +x2 residual, write d_out channel-first (fc2)
// ---------------------------------------------------------------------------
struct GemmArgs {
    const f16* A; const f16* Bw;
    int lda; int Kdim; int Nld;
    const float* bias;
    f16* outH;
    float* outF;        // x2 (mode 1)
    const float* resid; // shortcut/x2
    float* outDst;      // d_out (mode 3)
};

__device__ inline float gelu_exact(float u) {
    return 0.5f * u * (1.0f + erff(u * 0.70710678118654752f));
}

template <int MODE>
__global__ __launch_bounds__(256) void gemm128(GemmArgs ga)
{
    __shared__ f16  As[2][128 * 40];
    __shared__ f16  Bs[2][128 * 40];
    __shared__ float Es[32 * 132];

    int m0 = blockIdx.x * 128;
    int n0 = blockIdx.y * 128;
    int tid = threadIdx.x;
    int wave = tid >> 5, lane = tid & 31;
    int wave_m = wave >> 2, wave_n = wave & 3;
    int lm = lane & 15, hi = lane >> 4;

    v8f acc[4][2] = {};

    // issue one 128x32 A-tile + 128x32 B-tile as 4 async b128 per thread
    auto issue = [&](int buf, int kk) {
#pragma unroll
        for (int rep = 0; rep < 2; ++rep) {
            int idx = tid + rep * 256;
            int row = idx >> 2, q = idx & 3;
            async_cp16(ga.A  + (size_t)(m0 + row) * ga.lda  + kk + q * 8,
                       &As[buf][row * 40 + q * 8]);
            async_cp16(ga.Bw + (size_t)(n0 + row) * ga.Kdim + kk + q * 8,
                       &Bs[buf][row * 40 + q * 8]);
        }
    };

    int nIter = ga.Kdim >> 5;
    issue(0, 0);
    for (int it = 0; it < nIter; ++it) {
        int buf = it & 1;
        if (it + 1 < nIter) {
            issue(buf ^ 1, (it + 1) * 32);  // prefetch next tile into other buffer
            wait_async<4>();                // retire current tile, keep next in flight
        } else {
            wait_async<0>();
        }
        __syncthreads();

        v16h a[4], bf[2];
#pragma unroll
        for (int mt = 0; mt < 4; ++mt) {
            const f16* ap = &As[buf][(wave_m * 64 + mt * 16 + lm) * 40];
            h8 lo = *(const h8*)(ap + hi * 8);
            h8 hv = *(const h8*)(ap + 16 + hi * 8);
#pragma unroll
            for (int j = 0; j < 8; ++j) { a[mt][j] = lo[j]; a[mt][8 + j] = hv[j]; }
        }
#pragma unroll
        for (int nt = 0; nt < 2; ++nt) {
            const f16* bp = &Bs[buf][(wave_n * 32 + nt * 16 + lm) * 40] + hi * 8;
            h8 lo = *(const h8*)bp;
            h8 hv = *(const h8*)(bp + 16);
#pragma unroll
            for (int j = 0; j < 8; ++j) { bf[nt][j] = lo[j]; bf[nt][8 + j] = hv[j]; }
        }
#pragma unroll
        for (int mt = 0; mt < 4; ++mt)
#pragma unroll
            for (int nt = 0; nt < 2; ++nt)
                acc[mt][nt] = __builtin_amdgcn_wmma_f32_16x16x32_f16(
                    false, a[mt], false, bf[nt], (short)0, acc[mt][nt], false, false);
        __syncthreads();   // buffer reuse fence (next-next issue targets this buf)
    }

    // epilogue: 4 chunks of 32 rows staged through LDS
#pragma unroll 1
    for (int c4 = 0; c4 < 4; ++c4) {
        int r0 = c4 * 32;
        __syncthreads();
        if (wave_m == (r0 >> 6)) {
            int mtb = (r0 & 32) >> 4;
#pragma unroll
            for (int mo = 0; mo < 2; ++mo) {
                int mt = mtb + mo;
#pragma unroll
                for (int nt = 0; nt < 2; ++nt)
#pragma unroll
                    for (int j = 0; j < 8; ++j) {
                        int rl = mt * 16 + hi * 8 + j - (r0 & 63);
                        Es[rl * 132 + wave_n * 32 + nt * 16 + lm] = acc[mt][nt][j];
                    }
            }
        }
        __syncthreads();

        if (MODE == 3) {
            int cl = tid >> 1, half = tid & 1;
            int gCol = n0 + cl;
            float bcol = ga.bias[gCol];
            int m = m0 + r0 + half * 16;
            int bb = m >> 16, hh = (m & 65535) >> 8, ww0 = m & 255;
            float* dst = ga.outDst + (((size_t)(bb * CH + gCol) * HH + hh) * WW_ + ww0);
            const float* rs = ga.resid + (size_t)m * CH + gCol;
#pragma unroll
            for (int i = 0; i < 16; ++i)
                dst[i] = Es[(half * 16 + i) * 132 + cl] + bcol + rs[(size_t)i * CH];
        } else {
            int rloc = tid >> 3;
            int seg = (tid & 7) * 16;
            int gRow = m0 + r0 + rloc;
            int gCol = n0 + seg;
            float vals[16];
#pragma unroll
            for (int i = 0; i < 16; ++i) vals[i] = Es[rloc * 132 + seg + i] + ga.bias[gCol + i];

            if (MODE == 0 || MODE == 2) {
                union { f16 h[16]; uint4 u[2]; } pk;
#pragma unroll
                for (int i = 0; i < 16; ++i)
                    pk.h[i] = (f16)(MODE == 2 ? gelu_exact(vals[i]) : vals[i]);
                uint4* dp = (uint4*)(ga.outH + (size_t)gRow * ga.Nld + gCol);
                dp[0] = pk.u[0]; dp[1] = pk.u[1];
            } else { // MODE 1: window reverse + unshift + residual
                int win = gRow >> 6, n = gRow & 63;
                int bb = win >> 10, wh = (win >> 5) & 31, wwi = win & 31;
                int hsr = wh * 8 + (n >> 3), wsr = wwi * 8 + (n & 7);
                int h = (hsr + SHIFTV) & 255, w = (wsr + SHIFTV) & 255;
                size_t t = (size_t)bb * HW + (size_t)h * WW_ + w;
                const float* rs = ga.resid + t * CH + gCol;
                float* dst = ga.outF + t * CH + gCol;
#pragma unroll
                for (int i = 0; i < 16; ++i) dst[i] = rs[i] + vals[i];
            }
        }
    }
}

// ---------------------------------------------------------------------------
// K3: windowed attention. block = 128 threads (4 waves); wave = one head.
// blockIdx.x = win*2 + headGroup. Dynamic LDS: 4 * 21504 bytes.
// V tile staged via async DMA, retired just before the P@V phase.
// ---------------------------------------------------------------------------
__device__ inline int reg1d(int v) { return v < (HH - WSZ) ? 0 : (v < (HH - SHIFTV) ? 1 : 2); }

__global__ __launch_bounds__(128) void k3_attn(
    const f16* __restrict__ Y, f16* __restrict__ O,
    const float* __restrict__ rel_table)
{
    extern __shared__ char smem[];
    int tid = threadIdx.x;
    int wave = tid >> 5, lane = tid & 31;
    int lm = lane & 15, hi = lane >> 4;
    int win  = blockIdx.x >> 1;
    int head = ((blockIdx.x & 1) << 2) + wave;
    int wh = (win >> 5) & 31, wwi = win & 31;

    float* Sbuf = (float*)(smem + wave * 21504);            // [64][68] f32
    f16*   Pbuf = (f16*)Sbuf;                               // row stride 136 halves
    f16*   Vbuf = (f16*)(smem + wave * 21504 + 17408);      // [64][32] f16

    const f16* Yw = Y + (size_t)win * 64 * 768;
    const f16* Qp = Yw + head * 32;
    const f16* Kp = Yw + 256 + head * 32;
    const f16* Vp = Yw + 512 + head * 32;

    // stage V asynchronously (overlaps with the whole S/softmax phase)
#pragma unroll
    for (int r = lane; r < 64; r += 32) {
#pragma unroll
        for (int q = 0; q < 4; ++q)
            async_cp16(Vp + (size_t)r * 768 + q * 8, Vbuf + r * 32 + q * 8);
    }

    // ---- S = q @ k^T (16 WMMAs) ----
    v8f s[4][4] = {};
    v16h bf[4];
#pragma unroll
    for (int nt = 0; nt < 4; ++nt) {
        const f16* bp = Kp + (size_t)(nt * 16 + lm) * 768 + hi * 8;
        h8 lo = *(const h8*)bp;
        h8 hv = *(const h8*)(bp + 16);
#pragma unroll
        for (int j = 0; j < 8; ++j) { bf[nt][j] = lo[j]; bf[nt][8 + j] = hv[j]; }
    }
#pragma unroll
    for (int mt = 0; mt < 4; ++mt) {
        v16h a;
        const f16* ap = Qp + (size_t)(mt * 16 + lm) * 768;
        h8 lo = *(const h8*)(ap + hi * 8);
        h8 hv = *(const h8*)(ap + 16 + hi * 8);
#pragma unroll
        for (int j = 0; j < 8; ++j) { a[j] = lo[j]; a[8 + j] = hv[j]; }
#pragma unroll
        for (int nt = 0; nt < 4; ++nt)
            s[mt][nt] = __builtin_amdgcn_wmma_f32_16x16x32_f16(
                false, a, false, bf[nt], (short)0, s[mt][nt], false, false);
    }

    // dump S
#pragma unroll
    for (int mt = 0; mt < 4; ++mt)
#pragma unroll
        for (int nt = 0; nt < 4; ++nt)
#pragma unroll
            for (int j = 0; j < 8; ++j)
                Sbuf[(mt * 16 + hi * 8 + j) * 68 + nt * 16 + lm] = s[mt][nt][j];
    __syncthreads();

    // ---- bias + mask + softmax (2 rows per thread), write P f16 in place ----
#pragma unroll 1
    for (int rr = 0; rr < 2; ++rr) {
        int row = (lane << 1) + rr;
        int yi = row >> 3, xi = row & 7;
        int regR = reg1d(wh * 8 + yi) * 3 + reg1d(wwi * 8 + xi);
        float* srow = Sbuf + row * 68;
        float vals[64];
        float mx = -1e30f;
#pragma unroll
        for (int j = 0; j < 64; ++j) {
            int yj = j >> 3, xj = j & 7;
            int idx = (yi - yj + 7) * 15 + (xi - xj + 7);
            float bias = rel_table[idx * HEADS + head];
            int regC = reg1d(wh * 8 + yj) * 3 + reg1d(wwi * 8 + xj);
            float mk = (regR != regC) ? -100.0f : 0.0f;
            float v = srow[j] * ATTN_SCALE + bias + mk;
            vals[j] = v;
            mx = fmaxf(mx, v);
        }
        float sum = 0.f;
#pragma unroll
        for (int j = 0; j < 64; ++j) { vals[j] = expf(vals[j] - mx); sum += vals[j]; }
        float inv = 1.0f / sum;
        f16* prow = Pbuf + row * 136;
#pragma unroll
        for (int j = 0; j < 64; ++j) prow[j] = (f16)(vals[j] * inv);
    }
    wait_async<0>();   // V tile resident (own-wave DMA)
    __syncthreads();

    // ---- O = P @ V (16 WMMAs) ----
    v8f o[4][2] = {};
#pragma unroll
    for (int ka = 0; ka < 2; ++ka) {
        v16h vb[2];
#pragma unroll
        for (int nd = 0; nd < 2; ++nd)
#pragma unroll
            for (int j = 0; j < 16; ++j)
                vb[nd][j] = Vbuf[(ka * 32 + hi * 16 + j) * 32 + nd * 16 + lm];
#pragma unroll
        for (int mt = 0; mt < 4; ++mt) {
            v16h pa;
            const f16* pp = Pbuf + (size_t)(mt * 16 + lm) * 136 + ka * 32 + hi * 16;
            h8 lo = *(const h8*)pp;
            h8 hv = *(const h8*)(pp + 8);
#pragma unroll
            for (int j = 0; j < 8; ++j) { pa[j] = lo[j]; pa[8 + j] = hv[j]; }
#pragma unroll
            for (int nd = 0; nd < 2; ++nd)
                o[mt][nd] = __builtin_amdgcn_wmma_f32_16x16x32_f16(
                    false, pa, false, vb[nd], (short)0, o[mt][nd], false, false);
        }
    }
#pragma unroll
    for (int mt = 0; mt < 4; ++mt)
#pragma unroll
        for (int nd = 0; nd < 2; ++nd)
#pragma unroll
            for (int j = 0; j < 8; ++j) {
                int n = mt * 16 + hi * 8 + j;
                int d = nd * 16 + lm;
                O[((size_t)win * 64 + n) * CH + head * 32 + d] = (f16)o[mt][nd][j];
            }
}

// ---------------------------------------------------------------------------
// K5: LN2 (token-major x2 f32 -> x2n f16). One wave32 per token.
// ---------------------------------------------------------------------------
__global__ __launch_bounds__(256) void k5_ln2(
    const float* __restrict__ x2, const float* __restrict__ nw,
    const float* __restrict__ nb, f16* __restrict__ out)
{
    int wave = threadIdx.x >> 5, lane = threadIdx.x & 31;
    size_t t = (size_t)blockIdx.x * 8 + wave;
    const float* p = x2 + t * CH;
    float4 r0 = ((const float4*)p)[lane];
    float4 r1 = ((const float4*)p)[32 + lane];
    float sum = r0.x + r0.y + r0.z + r0.w + r1.x + r1.y + r1.z + r1.w;
    float sq  = r0.x*r0.x + r0.y*r0.y + r0.z*r0.z + r0.w*r0.w
              + r1.x*r1.x + r1.y*r1.y + r1.z*r1.z + r1.w*r1.w;
#pragma unroll
    for (int m = 1; m < 32; m <<= 1) {
        sum += __shfl_xor(sum, m, 32);
        sq  += __shfl_xor(sq,  m, 32);
    }
    float mu = sum * (1.0f / 256.0f);
    float var = sq * (1.0f / 256.0f) - mu * mu;
    float rs = rsqrtf(var + EPSV);
    int c0 = lane * 4, c1 = 128 + lane * 4;
    float a0[4] = { r0.x, r0.y, r0.z, r0.w };
    float a1[4] = { r1.x, r1.y, r1.z, r1.w };
#pragma unroll
    for (int i = 0; i < 4; ++i) {
        out[t * CH + c0 + i] = (f16)((a0[i] - mu) * rs * nw[c0 + i] + nb[c0 + i]);
        out[t * CH + c1 + i] = (f16)((a1[i] - mu) * rs * nw[c1 + i] + nb[c1 + i]);
    }
}

// ---------------------------------------------------------------------------
// Launch
// ---------------------------------------------------------------------------
extern "C" void kernel_launch(void* const* d_in, const int* in_sizes, int n_in,
                              void* d_out, int out_size, void* d_ws, size_t ws_size,
                              hipStream_t stream)
{
    const float* x        = (const float*)d_in[0];
    const float* norm1_w  = (const float*)d_in[1];
    const float* norm1_b  = (const float*)d_in[2];
    const float* qkv_w    = (const float*)d_in[3];
    const float* qkv_b    = (const float*)d_in[4];
    const float* proj_w   = (const float*)d_in[5];
    const float* proj_b   = (const float*)d_in[6];
    const float* relt     = (const float*)d_in[7];
    const float* norm2_w  = (const float*)d_in[8];
    const float* norm2_b  = (const float*)d_in[9];
    const float* fc1_w    = (const float*)d_in[10];
    const float* fc1_b    = (const float*)d_in[11];
    const float* fc2_w    = (const float*)d_in[12];
    const float* fc2_b    = (const float*)d_in[13];

    char* ws = (char*)d_ws;
    const size_t MB = 1ull << 20;
    f16*   xw   = (f16*)ws;                  // 128MB (reused as O, then x2n)
    f16*   qkvY = (f16*)(ws + 128 * MB);     // 384MB (region reused as 512MB hidden)
    f16*   hid  = (f16*)(ws + 128 * MB);
    float* x2   = (float*)(ws + 640 * MB);   // 256MB shortcut -> x2 (in-place residual)
    f16*   wth  = (f16*)(ws + 896 * MB);     // f16 weights ~1.5MB
    f16* qkvWh = wth;
    f16* projWh = qkvWh + 768 * 256;
    f16* fc1Wh  = projWh + 256 * 256;
    f16* fc2Wh  = fc1Wh + 1024 * 256;
    f16* O   = xw;
    f16* x2n = xw;

    // K0: weight conversion
    k_cvt<<<(768 * 256 + 255) / 256, 256, 0, stream>>>(qkv_w, qkvWh, 768 * 256);
    k_cvt<<<(256 * 256 + 255) / 256, 256, 0, stream>>>(proj_w, projWh, 256 * 256);
    k_cvt<<<(1024 * 256 + 255) / 256, 256, 0, stream>>>(fc1_w, fc1Wh, 1024 * 256);
    k_cvt<<<(256 * 1024 + 255) / 256, 256, 0, stream>>>(fc2_w, fc2Wh, 256 * 1024);

    // K1: LN1 + shift + window partition (+ shortcut transpose)
    k1_ln_shift<<<8192, 256, 0, stream>>>(x, norm1_w, norm1_b, x2, xw);

    // K2: QKV GEMM (M=262144, K=256, N=768)
    {
        GemmArgs ga = {};
        ga.A = xw; ga.Bw = qkvWh; ga.lda = 256; ga.Kdim = 256; ga.Nld = 768;
        ga.bias = qkv_b; ga.outH = qkvY;
        gemm128<0><<<dim3(NTOK / 128, 768 / 128), 256, 0, stream>>>(ga);
    }

    // K3: window attention (4096 windows * 8 heads; 4 heads per block)
    k3_attn<<<8192, 128, 4 * 21504, stream>>>(qkvY, O, relt);

    // K4: proj GEMM + window reverse/unshift + residual -> x2
    {
        GemmArgs ga = {};
        ga.A = O; ga.Bw = projWh; ga.lda = 256; ga.Kdim = 256; ga.Nld = 256;
        ga.bias = proj_b; ga.outF = x2; ga.resid = x2;
        gemm128<1><<<dim3(NTOK / 128, 256 / 128), 256, 0, stream>>>(ga);
    }

    // K5: LN2
    k5_ln2<<<NTOK / 8, 256, 0, stream>>>(x2, norm2_w, norm2_b, x2n);

    // K6: fc1 GEMM + GELU (M=262144, K=256, N=1024)
    {
        GemmArgs ga = {};
        ga.A = x2n; ga.Bw = fc1Wh; ga.lda = 256; ga.Kdim = 256; ga.Nld = 1024;
        ga.bias = fc1_b; ga.outH = hid;
        gemm128<2><<<dim3(NTOK / 128, 1024 / 128), 256, 0, stream>>>(ga);
    }

    // K7: fc2 GEMM + residual + channel-first transpose to d_out
    {
        GemmArgs ga = {};
        ga.A = hid; ga.Bw = fc2Wh; ga.lda = 1024; ga.Kdim = 1024; ga.Nld = 256;
        ga.bias = fc2_b; ga.resid = x2; ga.outDst = (float*)d_out;
        gemm128<3><<<dim3(NTOK / 128, 256 / 128), 256, 0, stream>>>(ga);
    }
}
